// MultiHeadAttention_3126736191599
// MI455X (gfx1250) — matrix operand order repaired and checked by reading, not compiled, not run on hardware
//
#include <hip/hip_runtime.h>
#include <hip/hip_bf16.h>

#define H 8
#define DIN 128
#define DEMB 128
#define DK 16
#define DV 16
#define BB 128
#define NN 256
#define NORMK 0.25f   /* 1/sqrt(D_K=16) */

typedef __attribute__((ext_vector_type(16))) _Float16 v16h;
typedef __attribute__((ext_vector_type(8)))  float    v8f;

__device__ __forceinline__ v8f wmma16(v16h a, v16h b, v8f c) {
  // D = A(16x32 f16) * B(32x16 f16) + C(16x16 f32)
  return __builtin_amdgcn_wmma_f32_16x16x32_f16(
      /*neg_a=*/false, a, /*neg_b=*/false, b,
      /*c_mod=*/(short)0, c, /*reuse_a=*/false, /*reuse_b=*/false);
}

__device__ __forceinline__ v16h zero16h() {
  v16h z;
#pragma unroll
  for (int i = 0; i < 16; ++i) z[i] = (_Float16)0.f;
  return z;
}

// LDS byte offset of a __shared__ object: LDS aperture maps addr[31:0] -> LDS offset.
__device__ __forceinline__ unsigned lds_off(const void* p) {
  return (unsigned)(size_t)p;
}

// ---------------------------------------------------------------------------
// Kernel 0: repack weights to WMMA-friendly f16 layouts.
// ---------------------------------------------------------------------------
__global__ void prep_weights(const float* __restrict__ Wq, const float* __restrict__ Wk,
                             const float* __restrict__ Wv, const float* __restrict__ Wo,
                             _Float16* __restrict__ WtQ, _Float16* __restrict__ WtK,
                             _Float16* __restrict__ WtV, _Float16* __restrict__ WoT) {
  int i = blockIdx.x * blockDim.x + threadIdx.x;
  if (i >= 128 * 128) return;
  int c = i >> 7, d = i & 127;
  int h = c >> 4, dk = c & 15;
  WtQ[c * 128 + d] = (_Float16)Wq[(h * DIN + d) * DK + dk];
  WtK[c * 128 + d] = (_Float16)Wk[(h * DIN + d) * DK + dk];
  WtV[c * 128 + d] = (_Float16)Wv[(h * DIN + d) * DK + dk];
  int e = i & 127, hd = i >> 7;
  int h2 = hd >> 4, dv = hd & 15;
  WoT[(h2 * DEMB + e) * 16 + dv] = (_Float16)Wo[(h2 * DV + dv) * DEMB + e];
}

// ---------------------------------------------------------------------------
// Kernel 1: Q/K/V projection (unchanged; 12 WMMAs / wave / block).
// ---------------------------------------------------------------------------
__global__ __launch_bounds__(256) void proj_qkv(
    const float* __restrict__ qin, const float* __restrict__ hin,
    const _Float16* __restrict__ WtQ, const _Float16* __restrict__ WtK,
    const _Float16* __restrict__ WtV,
    _Float16* __restrict__ Qp, _Float16* __restrict__ Kp, _Float16* __restrict__ VpT) {
  int blk = blockIdx.x;
  int b = blk >> 4;
  int rbase = (blk & 15) * 16;
  int w = threadIdx.x >> 5;
  int lane = threadIdx.x & 31;
  int hl = lane >> 4, nn = lane & 15;

#pragma unroll
  for (int mat = 0; mat < 3; ++mat) {
    const float* src = (mat == 0) ? qin : hin;
    const _Float16* Wt = (mat == 0) ? WtQ : (mat == 1) ? WtK : WtV;
    const float* arow = src + ((size_t)b * NN + rbase + nn) * DIN;
    v8f acc = {};
#pragma unroll
    for (int kc = 0; kc < 4; ++kc) {
      int d0 = kc * 32;
      v16h a;
#pragma unroll
      for (int j = 0; j < 8; ++j) a[j] = (_Float16)arow[d0 + 16 * hl + j];
#pragma unroll
      for (int j = 0; j < 8; ++j) a[8 + j] = (_Float16)arow[d0 + 16 * hl + 8 + j];
      v16h bf = *(const v16h*)(Wt + ((size_t)(w * 16 + nn)) * DIN + d0 + 16 * hl);
      acc = wmma16(a, bf, acc);
    }
    int c = w * 16 + nn;
    if (mat == 2) {
      int h2 = c >> 4, dv = c & 15;
      _Float16* dst = VpT + (((size_t)b * H + h2) * 16 + dv) * NN;
#pragma unroll
      for (int v = 0; v < 8; ++v) dst[rbase + v + 8 * hl] = (_Float16)acc[v];
    } else {
      _Float16* dst = (mat == 0) ? Qp : Kp;
#pragma unroll
      for (int v = 0; v < 8; ++v)
        dst[((size_t)b * NN + rbase + v + 8 * hl) * (H * DK) + c] = (_Float16)acc[v];
    }
  }
}

// ---------------------------------------------------------------------------
// Kernel 2: fused edge-MLP (layers 2+3 on WMMA) + masked flash attention +
// output projection. Edge/mask tiles staged via global_load_async_to_lds_b128.
// One block = (b, 16-query tile); wave w = head w; 8 chunks of 32 keys.
// ---------------------------------------------------------------------------
__global__ __launch_bounds__(256) void attn_fused(
    const _Float16* __restrict__ Qp, const _Float16* __restrict__ Kp,
    const _Float16* __restrict__ VpT, const _Float16* __restrict__ WoT,
    const unsigned char* __restrict__ mask, const float* __restrict__ edge,
    const float* __restrict__ mw1, const float* __restrict__ mb1,
    const float* __restrict__ mw2, const float* __restrict__ mb2,
    const float* __restrict__ mw3, const float* __restrict__ mb3,
    float* __restrict__ out) {
  __shared__ float    s_bias[H][16][32];    // per-head bias, mask folded (-1e30)
  __shared__ _Float16 s_P[H][16][32];       // softmax probs (A-frag source)
  __shared__ _Float16 s_O[H][16][16];       // normalized head outputs
  __shared__ float    s_out[16][DEMB];      // cross-head reduction (ds_add_f32)
  __shared__ _Float16 s_H[512][16];         // MLP hidden (layer1 out, then layer2 out)
  __shared__ float    s_edge[16][32];       // async-staged edge tile
  __shared__ unsigned char s_mask8[512];    // async-staged mask tile
  __shared__ _Float16 s_W2T[16][16];        // [j][i] = mw2[i][j]
  __shared__ _Float16 s_W3T[16][16];        // [c][i] = mw3[i][c] (c>=8 -> 0)
  __shared__ float    s_w1[16], s_b1[16], s_b2[16], s_b3[8];

  int blk = blockIdx.x;
  int b = blk >> 4;
  int qbase = (blk & 15) * 16;
  int tid = threadIdx.x;
  int h = tid >> 5;
  int lane = tid & 31;
  int hl = lane >> 4, nn = lane & 15;

  if (tid < 16) { s_w1[tid] = mw1[tid]; s_b1[tid] = mb1[tid]; s_b2[tid] = mb2[tid]; }
  if (tid < 8)   s_b3[tid] = mb3[tid];
  { // transposed f16 MLP weights for B-fragments
    int j = tid & 15, i = tid >> 4;
    s_W2T[j][i] = (_Float16)mw2[i * 16 + j];
    s_W3T[j][i] = (j < 8) ? (_Float16)mw3[i * 8 + j] : (_Float16)0.f;
  }
#pragma unroll
  for (int i = 0; i < 8; ++i) (&s_out[0][0])[tid + 256 * i] = 0.f;
  __syncthreads();

  // Loop-invariant B-fragments for MLP layers 2 and 3 (K = 16 padded to 32).
  v16h bW2, bW3;
  if (hl == 0) {
    bW2 = *(const v16h*)(&s_W2T[nn][0]);
    bW3 = *(const v16h*)(&s_W3T[nn][0]);
  } else {
    bW2 = zero16h();
    bW3 = zero16h();
  }

  // Q A-fragment (d_k=16 zero-padded to K=32)
  v16h aQ;
  {
    const _Float16* qrow = Qp + ((size_t)b * NN + qbase + nn) * (H * DK) + h * 16;
#pragma unroll
    for (int j = 0; j < 8; ++j) aQ[j] = hl ? (_Float16)0.f : qrow[j];
#pragma unroll
    for (int j = 0; j < 8; ++j) aQ[8 + j] = hl ? (_Float16)0.f : qrow[8 + j];
  }

  float rm[8], rl[8];
  v8f acc = {};
#pragma unroll
  for (int v = 0; v < 8; ++v) { rm[v] = -3e38f; rl[v] = 0.f; }

  const unsigned ldsE = lds_off(&s_edge[0][0]);
  const unsigned ldsM = lds_off(&s_mask8[0]);

  for (int kchunk = 0; kchunk < NN / 32; ++kchunk) {
    int k0 = kchunk * 32;
    __syncthreads();  // (A) previous chunk's consumers of s_edge/s_H/s_bias done

    // ---- async stage edge (16x32 f32) + mask (16x32 u8) tiles into LDS ----
    if (h == 0) {
      int r = lane >> 3, sg = lane & 7;  // 4 rows x 8 segs of 16B per b128
#pragma unroll
      for (int i = 0; i < 4; ++i) {
        unsigned dE = ldsE + (unsigned)(((4 * i + r) * 32 + sg * 4) * 4);
        unsigned long long gE = (unsigned long long)(const void*)(
            edge + ((size_t)b * NN + qbase + 4 * i + r) * NN + k0 + sg * 4);
        asm volatile("global_load_async_to_lds_b128 %0, %1, off"
                     :: "v"(dE), "v"(gE) : "memory");
      }
      unsigned dM = ldsM + (unsigned)((lane >> 1) * 32 + (lane & 1) * 16);
      unsigned long long gM = (unsigned long long)(const void*)(
          mask + ((size_t)b * NN + qbase + (lane >> 1)) * NN + k0 + (lane & 1) * 16);
      asm volatile("global_load_async_to_lds_b128 %0, %1, off"
                   :: "v"(dM), "v"(gM) : "memory");
      asm volatile("s_wait_asynccnt 0" ::: "memory");
    }
    __syncthreads();  // (B) tiles visible

    // ---- edge-MLP layer 1 (rank-1, VALU): h1 = relu(x*w1+b1) -> s_H f16 ----
#pragma unroll
    for (int e2 = 0; e2 < 2; ++e2) {
      int ei = tid + 256 * e2;
      float x = (&s_edge[0][0])[ei];
      v16h h1v;
#pragma unroll
      for (int j = 0; j < 16; ++j) {
        float t = fmaf(x, s_w1[j], s_b1[j]);
        h1v[j] = (_Float16)(t > 0.f ? t : 0.f);
      }
      *(v16h*)(&s_H[ei][0]) = h1v;
    }
    __syncthreads();  // (C) h1 visible

    // ---- layers 2+3 on WMMA: wave handles 4 edge tiles of 16 ----
#pragma unroll
    for (int t = 0; t < 4; ++t) {
      int tile = h * 4 + t;
      int erow = tile * 16 + nn;
      v16h a1;
      {
        const _Float16* hr = &s_H[erow][0];
#pragma unroll
        for (int j = 0; j < 8; ++j) a1[j] = hl ? (_Float16)0.f : hr[j];
#pragma unroll
        for (int j = 0; j < 8; ++j) a1[8 + j] = hl ? (_Float16)0.f : hr[8 + j];
      }
      v8f c2 = {};
      c2 = wmma16(a1, bW2, c2);
      // bias + relu in C layout, write back (in place, same wave) as layer-3 A source
#pragma unroll
      for (int v = 0; v < 8; ++v) {
        float t2 = c2[v] + s_b2[nn];
        s_H[tile * 16 + v + 8 * hl][nn] = (_Float16)(t2 > 0.f ? t2 : 0.f);
      }
      v16h a2;
      {
        const _Float16* hr = &s_H[erow][0];
#pragma unroll
        for (int j = 0; j < 8; ++j) a2[j] = hl ? (_Float16)0.f : hr[j];
#pragma unroll
        for (int j = 0; j < 8; ++j) a2[8 + j] = hl ? (_Float16)0.f : hr[8 + j];
      }
      v8f c3 = {};
      c3 = wmma16(a2, bW3, c3);
      if (nn < 8) {  // col = head channel
#pragma unroll
        for (int v = 0; v < 8; ++v) {
          int eidx = tile * 16 + v + 8 * hl;
          float bv = s_mask8[eidx] ? -1e30f : (c3[v] + s_b3[nn]);
          s_bias[nn][eidx >> 5][eidx & 31] = bv;
        }
      }
    }
    __syncthreads();  // (D) bias visible

    // ---- S = Q K^T for 32 keys (two 16x16 tiles, d_k padded to 32) ----
    v16h bK0, bK1;
    if (hl == 0) {
      bK0 = *(const v16h*)(Kp + ((size_t)b * NN + k0 + nn) * (H * DK) + h * 16);
      bK1 = *(const v16h*)(Kp + ((size_t)b * NN + k0 + 16 + nn) * (H * DK) + h * 16);
    } else {
      bK0 = zero16h();
      bK1 = zero16h();
    }
    v8f S0 = {}, S1 = {};
    S0 = wmma16(aQ, bK0, S0);
    S1 = wmma16(aQ, bK1, S1);

    // ---- bias + online softmax ----
#pragma unroll
    for (int v = 0; v < 8; ++v) {
      int m = v + 8 * hl;
      S0[v] = fmaf(S0[v], NORMK, s_bias[h][m][nn]);
      S1[v] = fmaf(S1[v], NORMK, s_bias[h][m][nn + 16]);
      float mx = fmaxf(S0[v], S1[v]);
#pragma unroll
      for (int off = 8; off >= 1; off >>= 1) mx = fmaxf(mx, __shfl_xor(mx, off, 32));
      float nm = fmaxf(rm[v], mx);
      float scale = __expf(rm[v] - nm);
      rm[v] = nm;
      float p0 = __expf(S0[v] - nm);
      float p1 = __expf(S1[v] - nm);
      float ps = p0 + p1;
#pragma unroll
      for (int off = 8; off >= 1; off >>= 1) ps += __shfl_xor(ps, off, 32);
      rl[v] = rl[v] * scale + ps;
      acc[v] *= scale;
      s_P[h][m][nn]      = (_Float16)p0;
      s_P[h][m][nn + 16] = (_Float16)p1;
    }

    // ---- O += P (16x32) @ V (32x16), full K=32 ----
    v16h aP;
    {
      const _Float16* prow = &s_P[h][nn][0];
#pragma unroll
      for (int j = 0; j < 8; ++j) aP[j] = prow[16 * hl + j];
#pragma unroll
      for (int j = 0; j < 8; ++j) aP[8 + j] = prow[16 * hl + 8 + j];
    }
    v16h bV = *(const v16h*)(VpT + (((size_t)b * H + h) * 16 + nn) * NN + k0 + 16 * hl);
    acc = wmma16(aP, bV, acc);
  }

  // ---- normalize, then out += O_h @ Wo_h with ds_add_f32 reduction ----
#pragma unroll
  for (int v = 0; v < 8; ++v) {
    float inv = 1.f / rl[v];
    s_O[h][v + 8 * hl][nn] = (_Float16)(acc[v] * inv);
  }
  v16h aO;
  {
    const _Float16* orow = &s_O[h][nn][0];
#pragma unroll
    for (int j = 0; j < 8; ++j) aO[j] = hl ? (_Float16)0.f : orow[j];
#pragma unroll
    for (int j = 0; j < 8; ++j) aO[8 + j] = hl ? (_Float16)0.f : orow[8 + j];
  }
#pragma unroll
  for (int et = 0; et < 8; ++et) {
    v16h bW;
    if (hl == 0)
      bW = *(const v16h*)(WoT + ((size_t)h * DEMB + et * 16 + nn) * 16);
    else
      bW = zero16h();
    v8f c = {};
    c = wmma16(aO, bW, c);
#pragma unroll
    for (int v = 0; v < 8; ++v)
      atomicAdd(&s_out[v + 8 * hl][et * 16 + nn], c[v]);
  }
  __syncthreads();

  float* og = out + ((size_t)b * NN + qbase) * DEMB;
#pragma unroll
  for (int i = 0; i < 8; ++i) {
    int idx = tid + 256 * i;
    og[idx] = (&s_out[0][0])[idx];
  }
}

// ---------------------------------------------------------------------------
extern "C" void kernel_launch(void* const* d_in, const int* in_sizes, int n_in,
                              void* d_out, int out_size, void* d_ws, size_t ws_size,
                              hipStream_t stream) {
  (void)in_sizes; (void)n_in; (void)out_size; (void)ws_size;
  const float* q    = (const float*)d_in[0];
  const float* hmat = (const float*)d_in[1];
  const unsigned char* mask = (const unsigned char*)d_in[2];
  const float* edge = (const float*)d_in[3];
  const float* Wq = (const float*)d_in[4];
  const float* Wk = (const float*)d_in[5];
  const float* Wv = (const float*)d_in[6];
  const float* Wo = (const float*)d_in[7];
  const float* mw1 = (const float*)d_in[8];
  const float* mb1 = (const float*)d_in[9];
  const float* mw2 = (const float*)d_in[10];
  const float* mb2 = (const float*)d_in[11];
  const float* mw3 = (const float*)d_in[12];
  const float* mb3 = (const float*)d_in[13];
  float* out = (float*)d_out;

  char* ws = (char*)d_ws;
  const size_t W16 = (size_t)128 * 128 * sizeof(_Float16);        // 32 KB
  const size_t P16 = (size_t)BB * NN * H * DK * sizeof(_Float16); // 8 MB
  _Float16* WtQ = (_Float16*)(ws);
  _Float16* WtK = (_Float16*)(ws + W16);
  _Float16* WtV = (_Float16*)(ws + 2 * W16);
  _Float16* WoT = (_Float16*)(ws + 3 * W16);
  _Float16* Qp  = (_Float16*)(ws + 4 * W16);
  _Float16* Kp  = (_Float16*)(ws + 4 * W16 + P16);
  _Float16* VpT = (_Float16*)(ws + 4 * W16 + 2 * P16);

  prep_weights<<<64, 256, 0, stream>>>(Wq, Wk, Wv, Wo, WtQ, WtK, WtV, WoT);
  proj_qkv<<<BB * (NN / 16), 256, 0, stream>>>(q, hmat, WtQ, WtK, WtV, Qp, Kp, VpT);
  attn_fused<<<BB * (NN / 16), 256, 0, stream>>>(Qp, Kp, VpT, WoT, mask, edge,
                                                 mw1, mb1, mw2, mb2, mw3, mb3, out);
}
